// TANet_74586402062765
// MI455X (gfx1250) — matrix operand order repaired and checked by reading, not compiled, not run on hardware
//
#include <hip/hip_runtime.h>
#include <hip/hip_fp16.h>

typedef _Float16 h16;
typedef __attribute__((ext_vector_type(16))) _Float16 v16h;
typedef __attribute__((ext_vector_type(8)))  float    v8f;

// ---------------- WMMA helpers (gfx1250, wave32) ----------------
// A fragment: row-major [16][lda] tile, one 16x32 K-slab.
// ISA layout (16-bit A 16x32): lane<16 -> K 0..7,16..23 ; lane>=16 -> K 8..15,24..31
__device__ __forceinline__ v16h frag_a(const h16* __restrict__ A, int lda) {
  int lane = threadIdx.x & 31;
  const h16* row = A + (lane & 15) * lda + ((lane >> 4) << 3);
  v16h f;
#pragma unroll
  for (int j = 0; j < 8; ++j) f[j] = row[j];
#pragma unroll
  for (int j = 0; j < 8; ++j) f[8 + j] = row[16 + j];
  return f;
}
// B fragment from transposed storage Bt[N][ldb] (contiguous K):
// lane<16 -> K 0..7,16..23 of column n=lane ; lane>=16 -> K 8..15,24..31 of column n=lane-16
__device__ __forceinline__ v16h frag_bt(const h16* __restrict__ Bt, int ldb) {
  int lane = threadIdx.x & 31;
  const h16* col = Bt + (lane & 15) * ldb + ((lane >> 4) << 3);
  v16h f;
#pragma unroll
  for (int j = 0; j < 8; ++j) f[j] = col[j];
#pragma unroll
  for (int j = 0; j < 8; ++j) f[8 + j] = col[16 + j];
  return f;
}
__device__ __forceinline__ v8f wmma32(v16h a, v16h b, v8f c) {
  return __builtin_amdgcn_wmma_f32_16x16x32_f16(false, a, false, b, (short)0, c,
                                                false, false);
}
// C/D layout: lane -> n = lane&15 ; register j -> m = j + 8*(lane>>4)

// generic epilogue for one 16x16 tile
__device__ __forceinline__ void tile_store(v8f acc, int mt16, int nt16,
                                           const float* __restrict__ bias,
                                           float* __restrict__ Cf, h16* __restrict__ Ch,
                                           int ldc, int nvalid, int mvalid, int act) {
  int lane = threadIdx.x & 31;
  int n = nt16 + (lane & 15);
  if (n >= nvalid) return;
  float bb = bias ? bias[n] : 0.f;
  int mbase = mt16 + ((lane >> 4) << 3);
#pragma unroll
  for (int j = 0; j < 8; ++j) {
    int m = mbase + j;
    if (m >= mvalid) continue;
    float v = acc[j] + bb;
    if (act == 1) v = fmaxf(v, 0.f);
    else if (act == 2) v = 1.f / (1.f + __expf(-v));
    else if (act == 3) v = tanhf(v);
    if (Cf) Cf[m * ldc + n] = v;
    if (Ch) Ch[m * ldc + n] = (h16)v;
  }
}

// ---------------- weight convert: f32 [count][K][N] -> f16 [count][Np][Kp] ----
__global__ void conv_wt(const float* __restrict__ src, h16* __restrict__ dst,
                        int K, int N, int Kp, int Np, int count) {
  int total = count * Np * Kp;
  for (int i = blockIdx.x * 256 + threadIdx.x; i < total; i += gridDim.x * 256) {
    int kp = i % Kp; int r = i / Kp; int n = r % Np; int c = r / Np;
    float v = (kp < K && n < N) ? src[(c * K + kp) * N + n] : 0.f;
    dst[i] = (h16)v;
  }
}

// ---------------- adjacency (A_in == A_out in reference), padded 48x160 f16 ---
__global__ void build_ain(h16* __restrict__ AIN) {
  int i = blockIdx.x * 256 + threadIdx.x;
  if (i >= 48 * 160) return;
  int r = i / 160, m = i % 160;
  float v = 0.f;
  if (r < 34 && m < 136) {
    int e = m / 34, c = m % 34;
    bool hit = false;
    if (e == 0)
      hit = (c == r + 1 && r < 32 && r != 15 && r != 31) ||
            (r == c + 1 && c < 32 && c != 15 && c != 31);
    else if (e == 1)
      hit = (r + c == 15 && r < 16 && c < 16) ||
            (r + c == 47 && r >= 16 && r < 32 && c >= 16 && c < 32);
    else if (e == 2)
      hit = (c == 32 && r < 16) || (r == 32 && c < 16) ||
            (c == 33 && r >= 16 && r < 32) || (r == 33 && c >= 16 && c < 32);
    else
      hit = (r == 32 && c == 33) || (r == 33 && c == 32);
    v = hit ? 1.f : 0.f;
  }
  AIN[i] = (h16)v;
}

// ---------------- per-node point encoder -------------------------------
// one block per (b,node): 512 points, MLP 3->64->128->512, max-pool
__global__ __launch_bounds__(256) void encoder_node(
    const float* __restrict__ points,
    const h16* __restrict__ W1t, const h16* __restrict__ W2t, const h16* __restrict__ W3t,
    const float* __restrict__ b1, const float* __restrict__ s1, const float* __restrict__ t1,
    const float* __restrict__ b2, const float* __restrict__ s2, const float* __restrict__ t2,
    const float* __restrict__ b3, const float* __restrict__ s3, const float* __restrict__ t3,
    float* __restrict__ EMB, float* __restrict__ HG, h16* __restrict__ HGH) {
  int b = blockIdx.x >> 5, nn = blockIdx.x & 31;
  int tid = threadIdx.x, wave = tid >> 5, lane = tid & 31;
  __shared__ h16 sW1[64 * 32];
  __shared__ h16 sW2[128 * 64];
  __shared__ h16 sPa[64 * 32];
  __shared__ h16 sH1[64 * 64];
  __shared__ h16 sH2[64 * 128];
  const h16* w1 = W1t + nn * 2048;
  for (int i = tid; i < 2048; i += 256) sW1[i] = w1[i];
  const h16* w2 = W2t + nn * 8192;
  for (int i = tid; i < 8192; i += 256) sW2[i] = w2[i];
  const float *B1 = b1 + nn * 64, *S1 = s1 + nn * 64, *T1 = t1 + nn * 64;
  const float *B2 = b2 + nn * 128, *S2 = s2 + nn * 128, *T2 = t2 + nn * 128;
  const float *B3 = b3 + nn * 512, *S3 = s3 + nn * 512, *T3 = t3 + nn * 512;
  const h16* W3n = W3t + nn * 512 * 128;
  float mx[4][8];
#pragma unroll
  for (int q = 0; q < 4; ++q)
#pragma unroll
    for (int j = 0; j < 8; ++j) mx[q][j] = 0.f;  // ReLU outputs are >= 0
  const float* pbase = points + (long)(b * 32 + nn) * 512 * 3;
  __syncthreads();
  for (int ch = 0; ch < 8; ++ch) {  // 64 points per chunk
    for (int i = tid; i < 2048; i += 256) {
      int pp = i >> 5, k = i & 31;
      sPa[i] = (k < 3) ? (h16)pbase[(ch * 64 + pp) * 3 + k] : (h16)0.f;
    }
    __syncthreads();
    // layer1: 4x4 tiles, K=32 (padded from 3)
    for (int t = 0; t < 2; ++t) {
      int tt = wave * 2 + t; int mt = tt >> 2, ntl = tt & 3;
      v8f acc = {};
      acc = wmma32(frag_a(sPa + mt * 16 * 32, 32), frag_bt(sW1 + ntl * 16 * 32, 32), acc);
      int n = ntl * 16 + (lane & 15);
      float bb = B1[n], sc = S1[n], sh = T1[n];
      int mrow = mt * 16 + ((lane >> 4) << 3);
#pragma unroll
      for (int j = 0; j < 8; ++j)
        sH1[(mrow + j) * 64 + n] = (h16)fmaxf((acc[j] + bb) * sc + sh, 0.f);
    }
    __syncthreads();
    // layer2: 4x8 tiles, K=64
    for (int t = 0; t < 4; ++t) {
      int tt = wave * 4 + t; int mt = tt >> 3, ntl = tt & 7;
      v8f acc = {};
#pragma unroll
      for (int k = 0; k < 64; k += 32)
        acc = wmma32(frag_a(sH1 + mt * 16 * 64 + k, 64), frag_bt(sW2 + ntl * 16 * 64 + k, 64), acc);
      int n = ntl * 16 + (lane & 15);
      float bb = B2[n], sc = S2[n], sh = T2[n];
      int mrow = mt * 16 + ((lane >> 4) << 3);
#pragma unroll
      for (int j = 0; j < 8; ++j)
        sH2[(mrow + j) * 128 + n] = (h16)fmaxf((acc[j] + bb) * sc + sh, 0.f);
    }
    __syncthreads();
    // layer3: N=512, weights streamed from L2, fused max-pool
#pragma unroll
    for (int q = 0; q < 4; ++q) {
      int ntl = wave + 8 * q;
      const h16* bt3 = W3n + ntl * 16 * 128;
      int n = ntl * 16 + (lane & 15);
      float bb = B3[n], sc = S3[n], sh = T3[n];
      for (int mt = 0; mt < 4; ++mt) {
        v8f acc = {};
#pragma unroll
        for (int k = 0; k < 128; k += 32)
          acc = wmma32(frag_a(sH2 + mt * 16 * 128 + k, 128), frag_bt(bt3 + k, 128), acc);
#pragma unroll
        for (int j = 0; j < 8; ++j)
          mx[q][j] = fmaxf(mx[q][j], fmaxf((acc[j] + bb) * sc + sh, 0.f));
      }
    }
    __syncthreads();
  }
#pragma unroll
  for (int q = 0; q < 4; ++q) {
    float v = mx[q][0];
#pragma unroll
    for (int j = 1; j < 8; ++j) v = fmaxf(v, mx[q][j]);
    v = fmaxf(v, __shfl_xor(v, 16));
    if (lane < 16) {
      int col = (wave + 8 * q) * 16 + lane;
      int idx = (b * 34 + nn) * 512 + col;
      EMB[idx] = v; HG[idx] = v; HGH[idx] = (h16)v;
    }
  }
}

// ---------------- global point encoder (shared weights, 3->64->128->1024) ----
__global__ __launch_bounds__(256) void encoder_global(
    const float* __restrict__ points,
    const h16* __restrict__ W1t, const h16* __restrict__ W2t, const h16* __restrict__ W3t,
    const float* __restrict__ b1, const float* __restrict__ s1, const float* __restrict__ t1,
    const float* __restrict__ b2, const float* __restrict__ s2, const float* __restrict__ t2,
    const float* __restrict__ b3, const float* __restrict__ s3, const float* __restrict__ t3,
    float* __restrict__ GEMB) {
  int b = blockIdx.x >> 5, blk = blockIdx.x & 31;
  int tid = threadIdx.x, wave = tid >> 5, lane = tid & 31;
  __shared__ h16 sW1[64 * 32];
  __shared__ h16 sW2[128 * 64];
  __shared__ h16 sPa[64 * 32];
  __shared__ h16 sH1[64 * 64];
  __shared__ h16 sH2[64 * 128];
  for (int i = tid; i < 2048; i += 256) sW1[i] = W1t[i];
  for (int i = tid; i < 8192; i += 256) sW2[i] = W2t[i];
  float mx[8][8];
#pragma unroll
  for (int q = 0; q < 8; ++q)
#pragma unroll
    for (int j = 0; j < 8; ++j) mx[q][j] = 0.f;
  const float* pbase = points + ((long)b * 16384 + blk * 512) * 3;
  __syncthreads();
  for (int ch = 0; ch < 8; ++ch) {
    for (int i = tid; i < 2048; i += 256) {
      int pp = i >> 5, k = i & 31;
      sPa[i] = (k < 3) ? (h16)pbase[(ch * 64 + pp) * 3 + k] : (h16)0.f;
    }
    __syncthreads();
    for (int t = 0; t < 2; ++t) {
      int tt = wave * 2 + t; int mt = tt >> 2, ntl = tt & 3;
      v8f acc = {};
      acc = wmma32(frag_a(sPa + mt * 16 * 32, 32), frag_bt(sW1 + ntl * 16 * 32, 32), acc);
      int n = ntl * 16 + (lane & 15);
      float bb = b1[n], sc = s1[n], sh = t1[n];
      int mrow = mt * 16 + ((lane >> 4) << 3);
#pragma unroll
      for (int j = 0; j < 8; ++j)
        sH1[(mrow + j) * 64 + n] = (h16)fmaxf((acc[j] + bb) * sc + sh, 0.f);
    }
    __syncthreads();
    for (int t = 0; t < 4; ++t) {
      int tt = wave * 4 + t; int mt = tt >> 3, ntl = tt & 7;
      v8f acc = {};
#pragma unroll
      for (int k = 0; k < 64; k += 32)
        acc = wmma32(frag_a(sH1 + mt * 16 * 64 + k, 64), frag_bt(sW2 + ntl * 16 * 64 + k, 64), acc);
      int n = ntl * 16 + (lane & 15);
      float bb = b2[n], sc = s2[n], sh = t2[n];
      int mrow = mt * 16 + ((lane >> 4) << 3);
#pragma unroll
      for (int j = 0; j < 8; ++j)
        sH2[(mrow + j) * 128 + n] = (h16)fmaxf((acc[j] + bb) * sc + sh, 0.f);
    }
    __syncthreads();
#pragma unroll
    for (int q = 0; q < 8; ++q) {
      int ntl = wave + 8 * q;
      const h16* bt3 = W3t + ntl * 16 * 128;
      int n = ntl * 16 + (lane & 15);
      float bb = b3[n], sc = s3[n], sh = t3[n];
      for (int mt = 0; mt < 4; ++mt) {
        v8f acc = {};
#pragma unroll
        for (int k = 0; k < 128; k += 32)
          acc = wmma32(frag_a(sH2 + mt * 16 * 128 + k, 128), frag_bt(bt3 + k, 128), acc);
#pragma unroll
        for (int j = 0; j < 8; ++j)
          mx[q][j] = fmaxf(mx[q][j], fmaxf((acc[j] + bb) * sc + sh, 0.f));
      }
    }
    __syncthreads();
  }
#pragma unroll
  for (int q = 0; q < 8; ++q) {
    float v = mx[q][0];
#pragma unroll
    for (int j = 1; j < 8; ++j) v = fmaxf(v, mx[q][j]);
    v = fmaxf(v, __shfl_xor(v, 16));
    if (lane < 16) {
      int col = (wave + 8 * q) * 16 + lane;
      // values are >=0 (ReLU) so int-compare == float-compare; GEMB zero-init
      atomicMax((int*)&GEMB[b * 1024 + col], __float_as_int(v));
    }
  }
}

// ---------------- GGNN: hin/hout matmuls, 2x2 register-blocked ----------------
// HGH padded to 288 rows (rows >=272 zero). Output scattered transposed:
// HINH/HOUTH [b][d][e*34+node], K-contiguous for the aggregation B operand.
__global__ __launch_bounds__(32) void ggnn_inout2(
    const h16* __restrict__ HGH, const h16* __restrict__ WIN, const h16* __restrict__ WOUT,
    const float* __restrict__ bin, const float* __restrict__ bout,
    h16* __restrict__ HINH, h16* __restrict__ HOUTH) {
  int t = blockIdx.x;
  int ntb = t & 15; t >>= 4;
  int mtb = t % 9; t /= 9;
  int e = t & 3; int dir = t >> 2;
  const h16* A0 = HGH + (mtb * 32) * 512;
  const h16* A1 = A0 + 16 * 512;
  const h16* B0 = (dir ? WOUT : WIN) + (e * 512 + ntb * 32) * 512;
  const h16* B1 = B0 + 16 * 512;
  v8f a00 = {}, a01 = {}, a10 = {}, a11 = {};
  for (int k = 0; k < 512; k += 32) {
    __builtin_prefetch(B0 + k + 256, 0, 1);
    __builtin_prefetch(B1 + k + 256, 0, 1);
    v16h x0 = frag_a(A0 + k, 512), x1 = frag_a(A1 + k, 512);
    v16h y0 = frag_bt(B0 + k, 512), y1 = frag_bt(B1 + k, 512);
    a00 = wmma32(x0, y0, a00); a01 = wmma32(x0, y1, a01);
    a10 = wmma32(x1, y0, a10); a11 = wmma32(x1, y1, a11);
  }
  int lane = threadIdx.x;
  const float* bv = dir ? bout : bin;
  h16* D = dir ? HOUTH : HINH;
  v8f ac[4] = {a00, a01, a10, a11};
#pragma unroll
  for (int ti = 0; ti < 2; ++ti)
#pragma unroll
    for (int tj = 0; tj < 2; ++tj) {
      v8f acc = ac[ti * 2 + tj];
      int n = ntb * 32 + tj * 16 + (lane & 15);
      float bb = bv[e * 512 + n];
      int mbase = mtb * 32 + ti * 16 + ((lane >> 4) << 3);
#pragma unroll
      for (int j = 0; j < 8; ++j) {
        int m = mbase + j;  // m = b*34+node
        if (m < 272) {
          int bI = m / 34, nnI = m % 34;
          D[(bI * 512 + n) * 160 + e * 34 + nnI] = (h16)(acc[j] + bb);
        }
      }
    }
}

// ---------------- GGNN: adjacency aggregation as WMMA ---------------------
__global__ __launch_bounds__(32) void ggnn_agg(
    const h16* __restrict__ AIN, const h16* __restrict__ HINH,
    const h16* __restrict__ HOUTH, h16* __restrict__ AH) {
  int t = blockIdx.x;
  int ntl = t & 31; t >>= 5;
  int mt = t % 3; t /= 3;
  int dir = t & 1; int b = t >> 1;
  const h16* Bt = (dir ? HOUTH : HINH) + (b * 512 + ntl * 16) * 160;
  const h16* Ab = AIN + mt * 16 * 160;
  int lane = threadIdx.x;
  v8f acc = {};
#pragma unroll
  for (int k = 0; k < 160; k += 32)
    acc = wmma32(frag_a(Ab + k, 160), frag_bt(Bt + k, 160), acc);
  int n = ntl * 16 + (lane & 15);
  int mrow = mt * 16 + ((lane >> 4) << 3);
#pragma unroll
  for (int j = 0; j < 8; ++j) {
    int m = mrow + j;
    if (m < 34) AH[(b * 34 + m) * 1536 + dir * 512 + n] = (h16)acc[j];
  }
}

// ---------------- elementwise fills of AH block 3 -------------------------
__global__ void copy_hg_block(const float* __restrict__ HG, h16* __restrict__ AH) {
  int i = blockIdx.x * 256 + threadIdx.x;
  if (i < 272 * 512) AH[(i >> 9) * 1536 + 1024 + (i & 511)] = (h16)HG[i];
}
__global__ void rh_block(const float* __restrict__ R, const float* __restrict__ HG,
                         h16* __restrict__ AH) {
  int i = blockIdx.x * 256 + threadIdx.x;
  if (i < 272 * 512) AH[(i >> 9) * 1536 + 1024 + (i & 511)] = (h16)(R[i] * HG[i]);
}

// ---------------- generic GEMMs with fused epilogue ------------------------
// act: 0=none 1=relu 2=sigmoid 3=tanh
// 1x1 tile (edge cases / tiny N)
__global__ __launch_bounds__(32) void gemm_tile(
    const h16* __restrict__ A, int lda, const h16* __restrict__ Bt, int ldb,
    const float* __restrict__ bias, int K, int Ntiles,
    float* __restrict__ Cf, h16* __restrict__ Ch, int ldc, int nvalid, int mvalid, int act) {
  int mt = blockIdx.x / Ntiles, ntl = blockIdx.x % Ntiles;
  const h16* Ab = A + mt * 16 * lda;
  const h16* Bb = Bt + ntl * 16 * ldb;
  v8f acc = {};
  for (int k = 0; k < K; k += 32)
    acc = wmma32(frag_a(Ab + k, lda), frag_bt(Bb + k, ldb), acc);
  tile_store(acc, mt * 16, ntl * 16, bias, Cf, Ch, ldc, nvalid, mvalid, act);
}
// 2x2 register-blocked: 32x32 of C per wave, 4 independent WMMAs per K-step
__global__ __launch_bounds__(32) void gemm_tile2(
    const h16* __restrict__ A, int lda, const h16* __restrict__ Bt, int ldb,
    const float* __restrict__ bias, int K, int Nt2,
    float* __restrict__ Cf, h16* __restrict__ Ch, int ldc, int nvalid, int mvalid, int act) {
  int mtb = blockIdx.x / Nt2, ntb = blockIdx.x % Nt2;
  const h16* A0 = A + (mtb * 32) * lda;
  const h16* A1 = A0 + 16 * lda;
  const h16* B0 = Bt + (ntb * 32) * ldb;
  const h16* B1 = B0 + 16 * ldb;
  v8f a00 = {}, a01 = {}, a10 = {}, a11 = {};
  for (int k = 0; k < K; k += 32) {
    __builtin_prefetch(B0 + k + 256, 0, 1);
    __builtin_prefetch(B1 + k + 256, 0, 1);
    v16h x0 = frag_a(A0 + k, lda), x1 = frag_a(A1 + k, lda);
    v16h y0 = frag_bt(B0 + k, ldb), y1 = frag_bt(B1 + k, ldb);
    a00 = wmma32(x0, y0, a00); a01 = wmma32(x0, y1, a01);
    a10 = wmma32(x1, y0, a10); a11 = wmma32(x1, y1, a11);
  }
  tile_store(a00, mtb * 32,      ntb * 32,      bias, Cf, Ch, ldc, nvalid, mvalid, act);
  tile_store(a01, mtb * 32,      ntb * 32 + 16, bias, Cf, Ch, ldc, nvalid, mvalid, act);
  tile_store(a10, mtb * 32 + 16, ntb * 32,      bias, Cf, Ch, ldc, nvalid, mvalid, act);
  tile_store(a11, mtb * 32 + 16, ntb * 32 + 16, bias, Cf, Ch, ldc, nvalid, mvalid, act);
}

// ---------------- GGNN GRU update, 2x2 blocked: hG=(1-z)hG+z*tanh(..) -----
__global__ __launch_bounds__(32) void ggnn_update2(
    const h16* __restrict__ AH, const h16* __restrict__ WT,
    const float* __restrict__ btv, const float* __restrict__ Z,
    float* __restrict__ HG, h16* __restrict__ HGH) {
  int mtb = blockIdx.x >> 4, ntb = blockIdx.x & 15;
  const h16* A0 = AH + (mtb * 32) * 1536;
  const h16* A1 = A0 + 16 * 1536;
  const h16* B0 = WT + (ntb * 32) * 1536;
  const h16* B1 = B0 + 16 * 1536;
  v8f a00 = {}, a01 = {}, a10 = {}, a11 = {};
  for (int k = 0; k < 1536; k += 32) {
    __builtin_prefetch(B0 + k + 256, 0, 1);
    __builtin_prefetch(B1 + k + 256, 0, 1);
    v16h x0 = frag_a(A0 + k, 1536), x1 = frag_a(A1 + k, 1536);
    v16h y0 = frag_bt(B0 + k, 1536), y1 = frag_bt(B1 + k, 1536);
    a00 = wmma32(x0, y0, a00); a01 = wmma32(x0, y1, a01);
    a10 = wmma32(x1, y0, a10); a11 = wmma32(x1, y1, a11);
  }
  int lane = threadIdx.x;
  v8f ac[4] = {a00, a01, a10, a11};
#pragma unroll
  for (int ti = 0; ti < 2; ++ti)
#pragma unroll
    for (int tj = 0; tj < 2; ++tj) {
      v8f acc = ac[ti * 2 + tj];
      int n = ntb * 32 + tj * 16 + (lane & 15);
      float bb = btv[n];
      int mbase = mtb * 32 + ti * 16 + ((lane >> 4) << 3);
#pragma unroll
      for (int j = 0; j < 8; ++j) {
        int m = mbase + j;
        if (m < 272) {
          int idx = m * 512 + n;
          float hh = tanhf(acc[j] + bb);
          float z = Z[idx];
          float v = (1.f - z) * HG[idx] + z * hh;
          HG[idx] = v; HGH[idx] = (h16)v;
        }
      }
    }
}

// ---------------- LayerNorm + final embedding assembly (f16 [256][1632]) --
__global__ __launch_bounds__(128) void ln_emb(
    const float* __restrict__ EMB, const float* __restrict__ HG,
    const float* __restrict__ GEMB, const float* __restrict__ centroid,
    const float* __restrict__ lng, const float* __restrict__ lnb,
    h16* __restrict__ EMBF) {
  int r = blockIdx.x; int b = r >> 5, nn = r & 31;
  int tid = threadIdx.x;
  __shared__ float tr[512];
  __shared__ float red[128];
  const float* e = EMB + (b * 34 + nn) * 512;
  const float* hg = HG + (b * 34 + nn) * 512;
  float s = 0.f;
  for (int i = tid; i < 512; i += 128) { float t = e[i] + hg[i]; tr[i] = t; s += t; }
  red[tid] = s; __syncthreads();
  for (int o = 64; o > 0; o >>= 1) { if (tid < o) red[tid] += red[tid + o]; __syncthreads(); }
  float mu = red[0] / 512.f;
  __syncthreads();
  float vv = 0.f;
  for (int i = tid; i < 512; i += 128) { float d = tr[i] - mu; vv += d * d; }
  red[tid] = vv; __syncthreads();
  for (int o = 64; o > 0; o >>= 1) { if (tid < o) red[tid] += red[tid + o]; __syncthreads(); }
  float inv = rsqrtf(red[0] / 512.f + 1e-5f);
  h16* out = EMBF + r * 1632;
  for (int c = tid; c < 1632; c += 128) {
    float val;
    if (c < 1024) val = GEMB[b * 1024 + c];
    else if (c < 1536) { int j = c - 1024; val = (tr[j] - mu) * inv * lng[j] + lnb[j]; }
    else val = centroid[b * 96 + (c - 1536)];
    out[c] = (h16)val;
  }
}

// =============================== host ===================================
extern "C" void kernel_launch(void* const* d_in, const int* in_sizes, int n_in,
                              void* d_out, int out_size, void* d_ws, size_t ws_size,
                              hipStream_t stream) {
  (void)in_sizes; (void)n_in; (void)out_size; (void)ws_size;
  const float* centroid = (const float*)d_in[0];
  const float* points = (const float*)d_in[1];
  const float* P[42];
  for (int i = 0; i < 42; ++i) P[i] = (const float*)d_in[2 + i];
  enum {
    enc_W1, enc_b1, enc_s1, enc_t1, enc_W2, enc_b2, enc_s2, enc_t2,
    enc_W3, enc_b3, enc_s3, enc_t3,
    g_W1, g_b1, g_s1, g_t1, g_W2, g_b2, g_s2, g_t2, g_W3, g_b3, g_s3, g_t3,
    gg_Win, gg_bin, gg_Wout, gg_bout, gg_Wr, gg_br, gg_Wz, gg_bz, gg_Wt, gg_bt,
    lng_, lnb_, r_W1, r_b1, r_W2, r_b2, r_W3, r_b3
  };

  char* wp = (char*)d_ws;
  auto take = [&](size_t bytes) -> void* {
    void* r = (void*)wp;
    wp += (bytes + 255) & ~(size_t)255;
    return r;
  };
  // f16 transposed weights [N][Kpad]
  h16* W_ENC1 = (h16*)take((size_t)32 * 64 * 32 * 2);
  h16* W_ENC2 = (h16*)take((size_t)32 * 128 * 64 * 2);
  h16* W_ENC3 = (h16*)take((size_t)32 * 512 * 128 * 2);
  h16* W_G1b  = (h16*)take((size_t)64 * 32 * 2);
  h16* W_G2b  = (h16*)take((size_t)128 * 64 * 2);
  h16* W_G3b  = (h16*)take((size_t)1024 * 128 * 2);
  h16* W_INb  = (h16*)take((size_t)4 * 512 * 512 * 2);
  h16* W_OUTb = (h16*)take((size_t)4 * 512 * 512 * 2);
  h16* W_Rb   = (h16*)take((size_t)512 * 1536 * 2);
  h16* W_Zb   = (h16*)take((size_t)512 * 1536 * 2);
  h16* W_Tb   = (h16*)take((size_t)512 * 1536 * 2);
  h16* W_R1b  = (h16*)take((size_t)512 * 1632 * 2);
  h16* W_R2b  = (h16*)take((size_t)256 * 512 * 2);
  h16* W_R3b  = (h16*)take((size_t)16 * 256 * 2);
  h16* AIN    = (h16*)take((size_t)48 * 160 * 2);
  // activations (M padded to 288 rows where 2x2 blocking reads past 272)
  float* EMB  = (float*)take((size_t)272 * 512 * 4);
  float* HG   = (float*)take((size_t)272 * 512 * 4);
  h16*   HGH  = (h16*)take((size_t)288 * 512 * 2);
  h16*   HINH = (h16*)take((size_t)8 * 512 * 160 * 2);
  h16*   HOUTH= (h16*)take((size_t)8 * 512 * 160 * 2);
  h16*   AH   = (h16*)take((size_t)288 * 1536 * 2);
  float* Rbuf = (float*)take((size_t)272 * 512 * 4);
  float* Zbuf = (float*)take((size_t)272 * 512 * 4);
  float* GEMB = (float*)take((size_t)8 * 1024 * 4);
  h16*   EMBF = (h16*)take((size_t)256 * 1632 * 2);
  h16*   H1   = (h16*)take((size_t)256 * 512 * 2);
  h16*   H2   = (h16*)take((size_t)256 * 256 * 2);

  auto cg = [](int total) { int g = (total + 255) / 256; return g > 8192 ? 8192 : g; };

  // zero-init (pad rows of HGH/AH/HINH/HOUTH, nodes 32/33, GEMB for atomicMax)
  hipMemsetAsync(EMB, 0, (size_t)272 * 512 * 4, stream);
  hipMemsetAsync(HG, 0, (size_t)272 * 512 * 4, stream);
  hipMemsetAsync(HGH, 0, (size_t)288 * 512 * 2, stream);
  hipMemsetAsync(HINH, 0, (size_t)8 * 512 * 160 * 2, stream);
  hipMemsetAsync(HOUTH, 0, (size_t)8 * 512 * 160 * 2, stream);
  hipMemsetAsync(AH, 0, (size_t)288 * 1536 * 2, stream);
  hipMemsetAsync(GEMB, 0, (size_t)8 * 1024 * 4, stream);

  // weight conversion/transposition to f16
  conv_wt<<<cg(32 * 64 * 32), 256, 0, stream>>>(P[enc_W1], W_ENC1, 3, 64, 32, 64, 32);
  conv_wt<<<cg(32 * 128 * 64), 256, 0, stream>>>(P[enc_W2], W_ENC2, 64, 128, 64, 128, 32);
  conv_wt<<<cg(32 * 512 * 128), 256, 0, stream>>>(P[enc_W3], W_ENC3, 128, 512, 128, 512, 32);
  conv_wt<<<cg(64 * 32), 256, 0, stream>>>(P[g_W1], W_G1b, 3, 64, 32, 64, 1);
  conv_wt<<<cg(128 * 64), 256, 0, stream>>>(P[g_W2], W_G2b, 64, 128, 64, 128, 1);
  conv_wt<<<cg(1024 * 128), 256, 0, stream>>>(P[g_W3], W_G3b, 128, 1024, 128, 1024, 1);
  conv_wt<<<cg(4 * 512 * 512), 256, 0, stream>>>(P[gg_Win], W_INb, 512, 512, 512, 512, 4);
  conv_wt<<<cg(4 * 512 * 512), 256, 0, stream>>>(P[gg_Wout], W_OUTb, 512, 512, 512, 512, 4);
  conv_wt<<<cg(512 * 1536), 256, 0, stream>>>(P[gg_Wr], W_Rb, 1536, 512, 1536, 512, 1);
  conv_wt<<<cg(512 * 1536), 256, 0, stream>>>(P[gg_Wz], W_Zb, 1536, 512, 1536, 512, 1);
  conv_wt<<<cg(512 * 1536), 256, 0, stream>>>(P[gg_Wt], W_Tb, 1536, 512, 1536, 512, 1);
  conv_wt<<<cg(512 * 1632), 256, 0, stream>>>(P[r_W1], W_R1b, 1632, 512, 1632, 512, 1);
  conv_wt<<<cg(256 * 512), 256, 0, stream>>>(P[r_W2], W_R2b, 512, 256, 512, 256, 1);
  conv_wt<<<cg(16 * 256), 256, 0, stream>>>(P[r_W3], W_R3b, 256, 9, 256, 16, 1);
  build_ain<<<30, 256, 0, stream>>>(AIN);

  // encoders
  encoder_node<<<256, 256, 0, stream>>>(points, W_ENC1, W_ENC2, W_ENC3,
      P[enc_b1], P[enc_s1], P[enc_t1], P[enc_b2], P[enc_s2], P[enc_t2],
      P[enc_b3], P[enc_s3], P[enc_t3], EMB, HG, HGH);
  encoder_global<<<256, 256, 0, stream>>>(points, W_G1b, W_G2b, W_G3b,
      P[g_b1], P[g_s1], P[g_t1], P[g_b2], P[g_s2], P[g_t2],
      P[g_b3], P[g_s3], P[g_t3], GEMB);

  // 5 GGNN propagation steps
  for (int s = 0; s < 5; ++s) {
    ggnn_inout2<<<9 * 16 * 8, 32, 0, stream>>>(HGH, W_INb, W_OUTb, P[gg_bin], P[gg_bout],
                                               HINH, HOUTH);
    ggnn_agg<<<8 * 3 * 32 * 2, 32, 0, stream>>>(AIN, HINH, HOUTH, AH);
    copy_hg_block<<<544, 256, 0, stream>>>(HG, AH);
    gemm_tile2<<<9 * 16, 32, 0, stream>>>(AH, 1536, W_Rb, 1536, P[gg_br], 1536, 16,
                                          Rbuf, (h16*)nullptr, 512, 512, 272, 2);
    gemm_tile2<<<9 * 16, 32, 0, stream>>>(AH, 1536, W_Zb, 1536, P[gg_bz], 1536, 16,
                                          Zbuf, (h16*)nullptr, 512, 512, 272, 2);
    rh_block<<<544, 256, 0, stream>>>(Rbuf, HG, AH);
    ggnn_update2<<<9 * 16, 32, 0, stream>>>(AH, W_Tb, P[gg_bt], Zbuf, HG, HGH);
  }

  // LayerNorm + concat(gemb, trans, centroid) -> f16 [256][1632]
  ln_emb<<<256, 128, 0, stream>>>(EMB, HG, GEMB, centroid, P[lng_], P[lnb_], EMBF);

  // regressor 1632 -> 512 -> 256 -> 9
  gemm_tile2<<<8 * 16, 32, 0, stream>>>(EMBF, 1632, W_R1b, 1632, P[r_b1], 1632, 16,
                                        (float*)nullptr, H1, 512, 512, 256, 1);
  gemm_tile2<<<8 * 8, 32, 0, stream>>>(H1, 512, W_R2b, 512, P[r_b2], 512, 8,
                                       (float*)nullptr, H2, 256, 256, 256, 1);
  gemm_tile<<<16, 32, 0, stream>>>(H2, 256, W_R3b, 256, P[r_b3], 256, 1,
                                   (float*)d_out, (h16*)nullptr, 9, 9, 256, 3);
}
